// GatedEGNOBlock_17291538333995
// MI455X (gfx1250) — matrix operand order repaired and checked by reading, not compiled, not run on hardware
//
#include <hip/hip_runtime.h>

#define T_   8
#define BN_  10000
#define C_   64
#define E_   160000
#define TBN  (T_*BN_)

typedef __attribute__((ext_vector_type(16))) __bf16 v16bf;
typedef __attribute__((ext_vector_type(8)))  float  v8f;

#define DEV static __device__ __forceinline__

DEV unsigned short f2bf(float f) {
  union { float f; unsigned u; } v; v.f = f;
  unsigned u = v.u;
  unsigned r = u + 0x7FFFu + ((u >> 16) & 1u);   // round-to-nearest-even
  return (unsigned short)(r >> 16);
}
DEV float bf2f(unsigned short h) {
  union { unsigned u; float f; } v; v.u = ((unsigned)h) << 16; return v.f;
}
DEV float silu_f(float x) { return x / (1.0f + __expf(-x)); }
DEV float sigm_f(float x) { return 1.0f / (1.0f + __expf(-x)); }

// Async global->LDS copy (CDNA5 GLOBAL_LOAD_ASYNC_TO_LDS_B128, ASYNCcnt-tracked).
// lds_dst: generic pointer derived from __shared__; low 32 bits are the LDS offset.
DEV void async_copy_b128(const void* gsrc, void* lds_dst) {
  unsigned lds_off = (unsigned)(unsigned long long)lds_dst;
  asm volatile("global_load_async_to_lds_b128 %0, %1, off"
               :: "v"(lds_off), "v"(gsrc) : "memory");
}
DEV void wait_async0() { asm volatile("s_wait_asynccnt 0x0" ::: "memory"); }

// A fragment (16x32 bf16): lane<16 row=ln holds K = kb+{0..15};
// lane>=16 same row, K = kb+16+{0..15} (contiguous, matches packed B). Arow = lane's row base.
DEV v16bf a_frag(const unsigned short* Arow, int kb, int half) {
  const unsigned short* p = Arow + kb + half * 16;
  v16bf r;
  __builtin_memcpy(&r, p, 16);                   // elems 0..7
  __builtin_memcpy((char*)&r + 16, p + 8, 16);   // elems 8..15 (K +8)
  return r;
}
// B fragment from fragment-major packed buffer: each lane's 16 bf16 contiguous (32B).
DEV v16bf b_frag(const unsigned short* Bp, int frag, int lane) {
  v16bf r;
  __builtin_memcpy(&r, Bp + ((size_t)frag * 32 + lane) * 16, 32);
  return r;
}
DEV v8f wmma_bf16(v16bf a, v16bf b, v8f c) {
  return __builtin_amdgcn_wmma_f32_16x16x32_bf16(false, a, false, b, (short)0, c,
                                                 false, false);
}

// ---------------- small prep kernels ----------------

// G[i][o][d] = (1/8)(wr0 + sum_m 2(wr_m cos(pi m d/4) + wi_m sin(pi m d/4)))
__global__ void build_G_kernel(const float* __restrict__ wr,
                               const float* __restrict__ wi,
                               float* __restrict__ G) {
  int idx = blockIdx.x * 256 + threadIdx.x;
  if (idx >= 64 * 64 * 8) return;
  int d = idx & 7, o = (idx >> 3) & 63, i = idx >> 9;
  float s = wr[(i * 64 + o) * 3 + 0];
#pragma unroll
  for (int m = 1; m < 3; m++) {
    float ang = 0.78539816339744830962f * (float)(m * d);  // 2*pi*m*d/8
    s += 2.0f * (wr[(i * 64 + o) * 3 + m] * __cosf(ang) +
                 wi[(i * 64 + o) * 3 + m] * __sinf(ang));
  }
  G[idx] = s * 0.125f;
}

// Pack the 8 time-shift matrices Bt[t] (512x64) into WMMA-B fragment-major bf16.
__global__ void pack_Bt_kernel(const float* __restrict__ G,
                               unsigned short* __restrict__ Bt) {
  int idx = blockIdx.x * 256 + threadIdx.x;
  if (idx >= 8 * 512 * 64) return;
  int o = idx & 63, k = (idx >> 6) & 511, t = idx >> 15;
  int tau = k >> 6, i = k & 63;
  int d = (tau - t) & 7;
  float val = G[(i * 64 + o) * 8 + d];
  int kt = k >> 5, kr = k & 31, half = kr >> 4, e = kr & 15;
  int nt = o >> 4, lane = half * 16 + (o & 15);
  Bt[(size_t)t * 32768 + (((kt * 4 + nt) * 32 + lane) * 16 + e)] = f2bf(val);
}

// Pack a (K x 64) row-major f32 weight into fragment-major bf16 (K = 64 or 128).
__global__ void pack_w_kernel(const float* __restrict__ src,
                              unsigned short* __restrict__ dst, int K) {
  int idx = blockIdx.x * 256 + threadIdx.x;
  if (idx >= K * 64) return;
  int n = idx & 63, k = idx >> 6;
  int kt = k >> 5, kr = k & 31, half = kr >> 4, e = kr & 15;
  int nt = n >> 4, lane = half * 16 + (n & 15);
  dst[((kt * 4 + nt) * 32 + lane) * 16 + e] = f2bf(src[k * 64 + n]);
}

__global__ void zero4_kernel(float4* __restrict__ p, int n4) {
  int i = blockIdx.x * 256 + threadIdx.x;
  if (i < n4) p[i] = make_float4(0.f, 0.f, 0.f, 0.f);
}

__global__ void f2bf_kernel(const float* __restrict__ src,
                            unsigned short* __restrict__ dst, int n) {
  int i = blockIdx.x * 256 + threadIdx.x;
  if (i < n) dst[i] = f2bf(src[i]);
}

__global__ void d2_kernel(const float* __restrict__ x, const int* __restrict__ ei,
                          float* __restrict__ d2) {
  int e = blockIdx.x * 256 + threadIdx.x;
  if (e >= E_) return;
  int r = ei[e], c = ei[E_ + e];
  float dx = x[r * 3 + 0] - x[c * 3 + 0];
  float dy = x[r * 3 + 1] - x[c * 3 + 1];
  float dz = x[r * 3 + 2] - x[c * 3 + 2];
  d2[e] = dx * dx + dy * dy + dz * dz;
}

// Equivariant velocity time-conv (channels=1): per (node,axis) 8-tap circulant.
__global__ void vel_kernel(const float* __restrict__ vel,
                           const float* __restrict__ wvr,
                           const float* __restrict__ wvi,
                           float* __restrict__ out) {
  int idx = blockIdx.x * 256 + threadIdx.x;
  if (idx >= BN_ * 3) return;
  int b = idx / 3, ax = idx - b * 3;
  float g[8];
#pragma unroll
  for (int d = 0; d < 8; d++) {
    float s = wvr[0];
#pragma unroll
    for (int m = 1; m < 3; m++) {
      float ang = 0.78539816339744830962f * (float)(m * d);
      s += 2.0f * (wvr[m] * __cosf(ang) + wvi[m] * __sinf(ang));
    }
    g[d] = s * 0.125f;
  }
  float v[8];
#pragma unroll
  for (int t = 0; t < 8; t++) v[t] = vel[b * 24 + t * 3 + ax];
#pragma unroll
  for (int t = 0; t < 8; t++) {
    float s = 0.f;
#pragma unroll
    for (int tau = 0; tau < 8; tau++) s += v[tau] * g[(tau - t) & 7];
    out[b * 24 + t * 3 + ax] = v[t] + s;
  }
}

// ---------------- TimeConv GEMM: (64 nodes x 512) @ (512 x 64) per t ----------------
#define TC_STRIDE 520  // 1040B row stride: 16B-aligned, 16 distinct bank groups
__global__ __launch_bounds__(128) void time_conv_kernel(
    const float* __restrict__ h, const unsigned short* __restrict__ Bt,
    float* __restrict__ hnew, unsigned short* __restrict__ hbf) {
  __shared__ unsigned short A[64 * TC_STRIDE];
  int node0 = blockIdx.x * 64;
  int tid = threadIdx.x;
  // Stage A[b_local][tau*64+i] as bf16 (512 rows of 64 over 128 threads)
  for (int r = tid; r < 512; r += 128) {
    int bl = r >> 3, tau = r & 7;
    int b = node0 + bl;
    unsigned short* dst = A + bl * TC_STRIDE + tau * 64;
    if (b < BN_) {
      const float* src = h + ((size_t)tau * BN_ + b) * 64;
#pragma unroll
      for (int i = 0; i < 64; i += 4) {
        float4 f = *(const float4*)(src + i);
        dst[i] = f2bf(f.x); dst[i + 1] = f2bf(f.y);
        dst[i + 2] = f2bf(f.z); dst[i + 3] = f2bf(f.w);
      }
    } else {
#pragma unroll
      for (int i = 0; i < 64; i++) dst[i] = 0;
    }
  }
  __syncthreads();
  int lane = tid & 31, wave = tid >> 5;
  int ln = lane & 15, half = lane >> 4;
  const unsigned short* Arow = A + (wave * 16 + ln) * TC_STRIDE;
  for (int t = 0; t < 8; t++) {
    v8f acc[4] = {};
    const unsigned short* Bp = Bt + (size_t)t * 32768;
#pragma unroll 4
    for (int kt = 0; kt < 16; kt++) {
      v16bf a = a_frag(Arow, kt * 32, half);
#pragma unroll
      for (int nt = 0; nt < 4; nt++)
        acc[nt] = wmma_bf16(a, b_frag(Bp, kt * 4 + nt, lane), acc[nt]);
    }
#pragma unroll
    for (int nt = 0; nt < 4; nt++) {
      int o = nt * 16 + ln;
#pragma unroll
      for (int e = 0; e < 8; e++) {
        int bl = wave * 16 + half * 8 + e;
        int b = node0 + bl;
        if (b < BN_) {
          size_t idx = ((size_t)t * BN_ + b) * 64 + o;
          float v = acc[nt][e];
          v = v > 0.f ? v : 0.2f * v;          // leaky_relu(0.2)
          float hn = h[idx] + v;
          hnew[idx] = hn;
          hbf[idx] = f2bf(hn);
        }
      }
    }
  }
}

// ---------------- Edge MLP: async gather -> GEMM(K=128) -> silu -> GEMM(K=64) ->
//                  silu -> gate -> atomic scatter-add ----------------
#define EG_STRIDE 88   // 176B: 16B-aligned, 16 distinct bank groups
__global__ __launch_bounds__(128) void edge_kernel(
    const unsigned short* __restrict__ hbf, const int* __restrict__ ei,
    const float* __restrict__ d2, const unsigned short* __restrict__ W1p,
    const float* __restrict__ We1full, const float* __restrict__ be1,
    const unsigned short* __restrict__ W2p, const float* __restrict__ be2,
    const float* __restrict__ Wg, const float* __restrict__ bg,
    float* __restrict__ agg) {
  __shared__ unsigned short Ah[128 * EG_STRIDE];  // rows 0..63 = hi, 64..127 = hj
  __shared__ unsigned short M2[64 * EG_STRIDE];
  __shared__ int   dstRow[64];
  __shared__ float d2s[64];
  int tile = blockIdx.x;
  int tsl = tile / 2500;                    // time slice
  int e0 = (tile - tsl * 2500) * 64;
  int tid = threadIdx.x;
  {
    int el = tid & 63, isJ = tid >> 6;
    int e = e0 + el;
    int node = isJ ? ei[E_ + e] : ei[e];
    if (!isJ) { dstRow[el] = node + tsl * BN_; d2s[el] = d2[e]; }
    const uint4* s = (const uint4*)(hbf + (size_t)(node + tsl * BN_) * 64);
    unsigned short* dd = Ah + (isJ * 64 + el) * EG_STRIDE;
#pragma unroll
    for (int q = 0; q < 8; q++)
      async_copy_b128(s + q, dd + q * 8);   // memory -> LDS, no VGPR round-trip
  }
  wait_async0();
  __syncthreads();
  int lane = tid & 31, wave = tid >> 5;
  int ln = lane & 15, half = lane >> 4;
  int mrow = wave * 16;
  // Layer 1: [hi|hj] (K=128) @ We1[0:128]
  v8f acc[4] = {};
  const unsigned short* Ai = Ah + (mrow + ln) * EG_STRIDE;
  const unsigned short* Aj = Ah + (64 + mrow + ln) * EG_STRIDE;
#pragma unroll
  for (int kt = 0; kt < 4; kt++) {
    const unsigned short* Ar = (kt < 2) ? Ai : Aj;
    v16bf a = a_frag(Ar, (kt & 1) * 32, half);
#pragma unroll
    for (int nt = 0; nt < 4; nt++)
      acc[nt] = wmma_bf16(a, b_frag(W1p, kt * 4 + nt, lane), acc[nt]);
  }
#pragma unroll
  for (int nt = 0; nt < 4; nt++) {
    int o = nt * 16 + ln;
    float wc = We1full[128 * 64 + o];   // d2 row of W_e1 (rank-1 epilogue)
    float bb = be1[o];
#pragma unroll
    for (int e = 0; e < 8; e++) {
      int el = mrow + half * 8 + e;
      float v = acc[nt][e] + d2s[el] * wc + bb;
      M2[el * EG_STRIDE + o] = f2bf(silu_f(v));
    }
  }
  __syncthreads();
  // Layer 2 (K=64)
  v8f a2[4] = {};
  const unsigned short* Mr = M2 + (mrow + ln) * EG_STRIDE;
#pragma unroll
  for (int kt = 0; kt < 2; kt++) {
    v16bf a = a_frag(Mr, kt * 32, half);
#pragma unroll
    for (int nt = 0; nt < 4; nt++)
      a2[nt] = wmma_bf16(a, b_frag(W2p, kt * 4 + nt, lane), a2[nt]);
  }
  float m2v[4][8], gp[8];
#pragma unroll
  for (int e = 0; e < 8; e++) gp[e] = 0.f;
#pragma unroll
  for (int nt = 0; nt < 4; nt++) {
    int o = nt * 16 + ln;
    float bb = be2[o], wg = Wg[o];
#pragma unroll
    for (int e = 0; e < 8; e++) {
      float v = silu_f(a2[nt][e] + bb);
      m2v[nt][e] = v;
      gp[e] += v * wg;
    }
  }
  // reduce gate dot across the 16 lanes that share each edge row
#pragma unroll
  for (int m = 1; m < 16; m <<= 1) {
#pragma unroll
    for (int e = 0; e < 8; e++) gp[e] += __shfl_xor(gp[e], m, 32);
  }
  float bgv = bg[0];
  float gate[8];
#pragma unroll
  for (int e = 0; e < 8; e++) gate[e] = sigm_f(gp[e] + bgv);
#pragma unroll
  for (int nt = 0; nt < 4; nt++) {
    int o = nt * 16 + ln;
#pragma unroll
    for (int e = 0; e < 8; e++) {
      int el = mrow + half * 8 + e;
      atomicAdd(&agg[(size_t)dstRow[el] * 64 + o], m2v[nt][e] * gate[e]);
    }
  }
}

// ---------------- Node MLP: [h|agg] (K=128) -> silu -> (K=64) -> residual ----------------
#define ND_STRIDE 136  // 272B: 16B-aligned, 16 distinct bank groups
__global__ __launch_bounds__(128) void node_kernel(
    const unsigned short* __restrict__ hbf, const unsigned short* __restrict__ aggbf,
    const float* __restrict__ hnew, const unsigned short* __restrict__ W1p,
    const float* __restrict__ b1, const unsigned short* __restrict__ W2p,
    const float* __restrict__ b2, float* __restrict__ out) {
  __shared__ unsigned short Ah[64 * ND_STRIDE];
  __shared__ unsigned short M2[64 * EG_STRIDE];
  int row0 = blockIdx.x * 64;
  int tid = threadIdx.x;
  {
    int el = tid & 63, part = tid >> 6;
    const unsigned short* src =
        (part ? aggbf : hbf) + (size_t)(row0 + el) * 64;
    const uint4* s = (const uint4*)src;
    unsigned short* dd = Ah + el * ND_STRIDE + part * 64;
#pragma unroll
    for (int q = 0; q < 8; q++)
      async_copy_b128(s + q, dd + q * 8);
  }
  wait_async0();
  __syncthreads();
  int lane = tid & 31, wave = tid >> 5;
  int ln = lane & 15, half = lane >> 4;
  int mrow = wave * 16;
  v8f acc[4] = {};
  const unsigned short* Ar = Ah + (mrow + ln) * ND_STRIDE;
#pragma unroll
  for (int kt = 0; kt < 4; kt++) {
    v16bf a = a_frag(Ar, kt * 32, half);
#pragma unroll
    for (int nt = 0; nt < 4; nt++)
      acc[nt] = wmma_bf16(a, b_frag(W1p, kt * 4 + nt, lane), acc[nt]);
  }
#pragma unroll
  for (int nt = 0; nt < 4; nt++) {
    int o = nt * 16 + ln;
    float bb = b1[o];
#pragma unroll
    for (int e = 0; e < 8; e++) {
      int el = mrow + half * 8 + e;
      M2[el * EG_STRIDE + o] = f2bf(silu_f(acc[nt][e] + bb));
    }
  }
  __syncthreads();
  v8f a2[4] = {};
  const unsigned short* Mr = M2 + (mrow + ln) * EG_STRIDE;
#pragma unroll
  for (int kt = 0; kt < 2; kt++) {
    v16bf a = a_frag(Mr, kt * 32, half);
#pragma unroll
    for (int nt = 0; nt < 4; nt++)
      a2[nt] = wmma_bf16(a, b_frag(W2p, kt * 4 + nt, lane), a2[nt]);
  }
#pragma unroll
  for (int nt = 0; nt < 4; nt++) {
    int o = nt * 16 + ln;
    float bb = b2[o];
#pragma unroll
    for (int e = 0; e < 8; e++) {
      int el = mrow + half * 8 + e;
      size_t idx = (size_t)(row0 + el) * 64 + o;
      out[idx] = hnew[idx] + a2[nt][e] + bb;
    }
  }
}

// ---------------- launch ----------------
extern "C" void kernel_launch(void* const* d_in, const int* in_sizes, int n_in,
                              void* d_out, int out_size, void* d_ws, size_t ws_size,
                              hipStream_t stream) {
  (void)in_sizes; (void)n_in; (void)out_size; (void)ws_size;
  const float* h   = (const float*)d_in[0];
  const float* x   = (const float*)d_in[1];
  const float* vel = (const float*)d_in[2];
  const int*   ei  = (const int*)d_in[3];
  const float* whr = (const float*)d_in[4];
  const float* whi = (const float*)d_in[5];
  const float* wvr = (const float*)d_in[6];
  const float* wvi = (const float*)d_in[7];
  const float* We1 = (const float*)d_in[8];
  const float* be1 = (const float*)d_in[9];
  const float* We2 = (const float*)d_in[10];
  const float* be2 = (const float*)d_in[11];
  const float* Wg  = (const float*)d_in[12];
  const float* bg  = (const float*)d_in[13];
  const float* Wn1 = (const float*)d_in[14];
  const float* bn1 = (const float*)d_in[15];
  const float* Wn2 = (const float*)d_in[16];
  const float* bn2 = (const float*)d_in[17];
  float* outH = (float*)d_out;                 // (T,BN,C)
  float* outV = outH + (size_t)TBN * 64;       // (BN,T,3)

  char* ws = (char*)d_ws;
  size_t off = 0;
  auto carve = [&](size_t bytes) -> void* {
    void* p = ws + off;
    off += (bytes + 255) & ~(size_t)255;
    return p;
  };
  float*          G     = (float*)carve(64 * 64 * 8 * 4);
  unsigned short* Bt    = (unsigned short*)carve((size_t)8 * 512 * 64 * 2);
  unsigned short* W1p   = (unsigned short*)carve(128 * 64 * 2);
  unsigned short* W2p   = (unsigned short*)carve(64 * 64 * 2);
  unsigned short* Wn1p  = (unsigned short*)carve(128 * 64 * 2);
  unsigned short* Wn2p  = (unsigned short*)carve(64 * 64 * 2);
  unsigned short* hbf   = (unsigned short*)carve((size_t)TBN * 64 * 2);
  unsigned short* aggbf = (unsigned short*)carve((size_t)TBN * 64 * 2);
  float*          aggf  = (float*)carve((size_t)TBN * 64 * 4);
  float*          hnew  = (float*)carve((size_t)TBN * 64 * 4);
  float*          d2    = (float*)carve((size_t)E_ * 4);

  build_G_kernel<<<(64 * 64 * 8 + 255) / 256, 256, 0, stream>>>(whr, whi, G);
  pack_Bt_kernel<<<(8 * 512 * 64 + 255) / 256, 256, 0, stream>>>(G, Bt);
  pack_w_kernel<<<(128 * 64 + 255) / 256, 256, 0, stream>>>(We1, W1p, 128);
  pack_w_kernel<<<(64 * 64 + 255) / 256, 256, 0, stream>>>(We2, W2p, 64);
  pack_w_kernel<<<(128 * 64 + 255) / 256, 256, 0, stream>>>(Wn1, Wn1p, 128);
  pack_w_kernel<<<(64 * 64 + 255) / 256, 256, 0, stream>>>(Wn2, Wn2p, 64);
  zero4_kernel<<<((TBN * 64 / 4) + 255) / 256, 256, 0, stream>>>((float4*)aggf,
                                                                 TBN * 64 / 4);
  d2_kernel<<<(E_ + 255) / 256, 256, 0, stream>>>(x, ei, d2);
  time_conv_kernel<<<(BN_ + 63) / 64, 128, 0, stream>>>(h, Bt, hnew, hbf);
  vel_kernel<<<(BN_ * 3 + 255) / 256, 256, 0, stream>>>(vel, wvr, wvi, outV);
  edge_kernel<<<(T_ * E_) / 64, 128, 0, stream>>>(hbf, ei, d2, W1p, We1, be1,
                                                  W2p, be2, Wg, bg, aggf);
  f2bf_kernel<<<(TBN * 64 + 255) / 256, 256, 0, stream>>>(aggf, aggbf, TBN * 64);
  node_kernel<<<TBN / 64, 128, 0, stream>>>(hbf, aggbf, hnew, Wn1p, bn1, Wn2p,
                                            bn2, outH);
}
